// PointTransformerEncoder_21912923144350
// MI455X (gfx1250) — matrix operand order repaired and checked by reading, not compiled, not run on hardware
//
#include <hip/hip_runtime.h>

// ---------------- CDNA5 WMMA types ----------------
typedef __attribute__((ext_vector_type(16))) __bf16 v16bf;
typedef __attribute__((ext_vector_type(8)))  __bf16 v8bf;
typedef __attribute__((ext_vector_type(8)))  float  v8f;

static constexpr int BB   = 2;
static constexpr int SS   = 8;
static constexpr int NN   = 1024;
static constexpr int FEAT = 6;
static constexpr int HID  = 64;
static constexpr int KNB  = 16;
static constexpr int NP   = BB * SS * NN;   // 16384 points
static constexpr float EPSV = 1e-6f;

__device__ inline __bf16 tobf(float f) {
    unsigned int u = __float_as_uint(f);
    u += 0x7FFFu + ((u >> 16) & 1u);           // round to nearest even
    unsigned short h = (unsigned short)(u >> 16);
    return __builtin_bit_cast(__bf16, h);
}

__device__ inline float wave_sum32(float v) {
#pragma unroll
    for (int off = 16; off > 0; off >>= 1) v += __shfl_xor(v, off, 32);
    return v;
}

// ---------------- kNN: per (b,s) slice, top-16 smallest dist ----------------
__global__ void __launch_bounds__(256) knn_kernel(const float* __restrict__ pts,
                                                  int* __restrict__ nbr) {
    __shared__ float sp[NN * 3];
    int grp  = blockIdx.x >> 2;          // (b*S+s)
    int base = grp * NN;
    for (int i = threadIdx.x; i < NN; i += 256) {
        const float* p = pts + (size_t)(base + i) * FEAT;
        sp[i * 3 + 0] = p[0]; sp[i * 3 + 1] = p[1]; sp[i * 3 + 2] = p[2];
    }
    __syncthreads();
    int i  = ((blockIdx.x & 3) << 8) + threadIdx.x;
    float px = sp[i * 3], py = sp[i * 3 + 1], pz = sp[i * 3 + 2];
    float bd[KNB]; int bi[KNB];
#pragma unroll
    for (int k = 0; k < KNB; ++k) { bd[k] = 3.4e38f; bi[k] = 0; }
    for (int j = 0; j < NN; ++j) {
        float dx = sp[j * 3] - px, dy = sp[j * 3 + 1] - py, dz = sp[j * 3 + 2] - pz;
        float d  = dx * dx + dy * dy + dz * dz;
        if (d < bd[KNB - 1]) {
            int k = KNB - 1;
            while (k > 0 && bd[k - 1] > d) { bd[k] = bd[k - 1]; bi[k] = bi[k - 1]; --k; }
            bd[k] = d; bi[k] = j;
        }
    }
    for (int k = 0; k < KNB; ++k) nbr[(size_t)(base + i) * KNB + k] = bi[k];
}

// ---------------- x = points @ W0 + b0 (6 -> 64), also emit bf16 ----------------
__global__ void __launch_bounds__(256) embed_kernel(const float* __restrict__ pts,
                                                    const float* __restrict__ W0,
                                                    const float* __restrict__ b0,
                                                    float* __restrict__ x,
                                                    __bf16* __restrict__ xb) {
    int idx = blockIdx.x * 256 + threadIdx.x;
    if (idx >= NP * HID) return;
    int i = idx >> 6, h = idx & 63;
    const float* p = pts + (size_t)i * FEAT;
    float a = b0[h];
#pragma unroll
    for (int f = 0; f < FEAT; ++f) a += p[f] * W0[f * HID + h];
    x[idx] = a;
    xb[idx] = tobf(a);
}

// ---------------- rel[i][k][:] = pos[nbr] - pos[i] ----------------
__global__ void __launch_bounds__(256) rel_kernel(const float* __restrict__ pts,
                                                  const int* __restrict__ nbr,
                                                  float* __restrict__ rel) {
    int idx = blockIdx.x * 256 + threadIdx.x;       // NP*KNB
    if (idx >= NP * KNB) return;
    int i = idx >> 4;
    int base = (i >> 10) << 10;
    int j = nbr[idx];
    const float* pi = pts + (size_t)i * FEAT;
    const float* pj = pts + (size_t)(base + j) * FEAT;
    rel[(size_t)idx * 3 + 0] = pj[0] - pi[0];
    rel[(size_t)idx * 3 + 1] = pj[1] - pi[1];
    rel[(size_t)idx * 3 + 2] = pj[2] - pi[2];
}

// ---------------- fp32 -> bf16 cast (activations, row-major kept) ----------------
__global__ void __launch_bounds__(256) cast_bf16_kernel(const float* __restrict__ in,
                                                        __bf16* __restrict__ out, int n) {
    int i = blockIdx.x * 256 + threadIdx.x;
    if (i < n) out[i] = tobf(in[i]);
}

// ---- weight cast+pack into WMMA B-fragment order:
//      Wp[(g*64+n)*16 + t] = bf16( W[(g*16+t)*64 + n] ),  g=K/16 group, n=col, t=K%16
//      -> each lane's B fragment is one contiguous 32-byte span.
__global__ void __launch_bounds__(256) pack_w_kernel(const float* __restrict__ W,
                                                     __bf16* __restrict__ Wp) {
    int o = blockIdx.x * 256 + threadIdx.x;          // 4096
    if (o >= HID * HID) return;
    int t = o & 15, n = (o >> 4) & 63, g = o >> 10;
    Wp[o] = tobf(W[(g * 16 + t) * HID + n]);
}

// ---------------- WMMA GEMM: C[M x 64] = A[M x 64](bf16) @ W(packed bf16) + bias ----
// One 16x16 output tile per wave; 4 waves/block cover 64 columns.
// K=64 -> two v_wmma_f32_16x16x32_bf16, f32 accumulation.
// A fragment: two b128 loads per chunk (contiguous per lane).
// B fragment: one contiguous 32B load per chunk (pre-packed).
__global__ void __launch_bounds__(128) gemm_bf16_kernel(const __bf16* __restrict__ A,
                                                        const __bf16* __restrict__ Wp,
                                                        const float* __restrict__ bias,
                                                        float* __restrict__ C) {
    int lane = threadIdx.x & 31;
    int wave = threadIdx.x >> 5;
    int row0 = blockIdx.x << 4;
    int col0 = wave << 4;
    int m  = lane & 15;          // A row within tile / B,C column within tile
    int hl = lane >> 4;          // lane-half select
    const __bf16* arow = A + (size_t)(row0 + m) * HID;
    int n = col0 + m;

    v8f acc = {};
#pragma unroll
    for (int c = 0; c < 2; ++c) {
        int kb = c * 32;
        // A 16x32 layout: lane holds K = kb + hl*16 .. +15 (contiguous, matches B)
        v8bf a_lo = *(const v8bf*)(arow + kb + hl * 16);
        v8bf a_hi = *(const v8bf*)(arow + kb + hl * 16 + 8);
        v16bf a = __builtin_shufflevector(a_lo, a_hi,
                    0, 1, 2, 3, 4, 5, 6, 7, 8, 9, 10, 11, 12, 13, 14, 15);
        // B 32x16 layout: lane holds column n, K = kb + hl*16 .. +15 (pre-packed contiguous)
        int g = c * 2 + hl;
        v16bf b = *(const v16bf*)(Wp + ((size_t)(g * 64 + n) << 4));
        acc = __builtin_amdgcn_wmma_f32_16x16x32_bf16(
            /*neg_a=*/false, a, /*neg_b=*/false, b,
            /*c_mod=*/(short)0, acc, /*reuse_a=*/false, /*reuse_b=*/false);
    }
    // C/D: VGPR r -> row = hl*8 + r, col = lane&15
#pragma unroll
    for (int r = 0; r < 8; ++r)
        C[(size_t)(row0 + hl * 8 + r) * HID + n] = acc[r] + bias[n];
}

// ---------------- pe = pos @ Wpe + bpe (3 -> 64) ----------------
__global__ void __launch_bounds__(256) pe_kernel(const float* __restrict__ pts,
                                                 const float* __restrict__ W,
                                                 const float* __restrict__ bias,
                                                 float* __restrict__ pe) {
    int idx = blockIdx.x * 256 + threadIdx.x;
    if (idx >= NP * HID) return;
    int i = idx >> 6, h = idx & 63;
    const float* p = pts + (size_t)i * FEAT;
    pe[idx] = bias[h] + p[0] * W[h] + p[1] * W[HID + h] + p[2] * W[2 * HID + h];
}

// ---------------- neighbor attention (vector attention, K=16) ----------------
__global__ void __launch_bounds__(64) attn_kernel(const float* __restrict__ q,
                                                  const float* __restrict__ kf,
                                                  const float* __restrict__ vf,
                                                  const float* __restrict__ pe,
                                                  const float* __restrict__ rel,
                                                  const int* __restrict__ nbr,
                                                  const float* __restrict__ Wpd,
                                                  const float* __restrict__ bpd,
                                                  const float* __restrict__ Wa,
                                                  const float* __restrict__ ba,
                                                  float* __restrict__ outb) {
    __shared__ float part[2];
    __shared__ float logits[KNB];
    __shared__ int   sj[KNB];
    __shared__ float srel[KNB * 3];
    int i = blockIdx.x, c = threadIdx.x;
    if (c < KNB) sj[c] = nbr[(size_t)i * KNB + c];
    if (c < KNB * 3) srel[c] = rel[(size_t)i * KNB * 3 + c];
    __syncthreads();
    int base = (i >> 10) << 10;
    float qn  = q[(size_t)i * HID + c] + pe[(size_t)i * HID + c];
    float wac = Wa[c];
    for (int kk = 0; kk < KNB; ++kk) {
        int jg = base + sj[kk];
        float pd = srel[kk * 3] * Wpd[c] + srel[kk * 3 + 1] * Wpd[HID + c]
                 + srel[kk * 3 + 2] * Wpd[2 * HID + c] + bpd[c];
        float feat = qn * (kf[(size_t)jg * HID + c] + pe[(size_t)jg * HID + c] + pd);
        float v = wave_sum32(feat * wac);
        if ((c & 31) == 0) part[c >> 5] = v;
        __syncthreads();
        if (c == 0) logits[kk] = part[0] + part[1] + ba[0];
        __syncthreads();
    }
    float mx = -3.4e38f;
#pragma unroll
    for (int kk = 0; kk < KNB; ++kk) mx = fmaxf(mx, logits[kk]);
    float w[KNB]; float se = 0.f;
#pragma unroll
    for (int kk = 0; kk < KNB; ++kk) { w[kk] = expf(logits[kk] - mx); se += w[kk]; }
    float inv = 1.f / se;
    float oc = 0.f;
#pragma unroll
    for (int kk = 0; kk < KNB; ++kk) oc += w[kk] * vf[(size_t)(base + sj[kk]) * HID + c];
    outb[(size_t)i * HID + c] = oc * inv;
}

// ---------------- x = LN(x + relu(o)) * g + b ; re-emit bf16 x ----------------
__global__ void __launch_bounds__(64) resln_kernel(float* __restrict__ x,
                                                   const float* __restrict__ o,
                                                   const float* __restrict__ g,
                                                   const float* __restrict__ bln,
                                                   __bf16* __restrict__ xb) {
    __shared__ float part[2];
    __shared__ float sm, sv;
    int i = blockIdx.x, c = threadIdx.x;
    float val = x[(size_t)i * HID + c] + fmaxf(o[(size_t)i * HID + c], 0.f);
    float v = wave_sum32(val);
    if ((c & 31) == 0) part[c >> 5] = v;
    __syncthreads();
    if (c == 0) sm = (part[0] + part[1]) * (1.f / 64.f);
    __syncthreads();
    float d = val - sm;
    v = wave_sum32(d * d);
    if ((c & 31) == 0) part[c >> 5] = v;
    __syncthreads();
    if (c == 0) sv = (part[0] + part[1]) * (1.f / 64.f);
    __syncthreads();
    float y = d * rsqrtf(sv + EPSV) * g[c] + bln[c];
    x[(size_t)i * HID + c] = y;
    xb[(size_t)i * HID + c] = tobf(y);
}

// ---------------- max over N per (b,s,channel) ----------------
__global__ void __launch_bounds__(64) maxpool_kernel(const float* __restrict__ x,
                                                     float* __restrict__ xp) {
    int g = blockIdx.x, c = threadIdx.x;
    float m = -3.4e38f;
    for (int n = 0; n < NN; ++n) m = fmaxf(m, x[((size_t)g * NN + n) * HID + c]);
    xp[g * HID + c] = m;
}

// ---------------- final tiny encoder over 16 tokens (2 batches x 8) ----------------
__global__ void __launch_bounds__(256) final_kernel(const float* __restrict__ xp,
        const float* Wk, const float* bk, const float* Wq, const float* bq,
        const float* Wv, const float* bv, const float* Wo1, const float* bo1,
        const float* Wo2, const float* bo2, const float* gf, const float* bff,
        float* __restrict__ out) {
    __shared__ float xs[16 * 64], ks[16 * 64], qs[16 * 64], vs[16 * 64], t1[16 * 64];
    __shared__ float aw[2 * 8 * 8];
    int tid = threadIdx.x;
    for (int idx = tid; idx < 1024; idx += 256) xs[idx] = xp[idx];
    __syncthreads();
    for (int idx = tid; idx < 1024; idx += 256) {
        int t = idx >> 6, h = idx & 63;
        float ak = bk[h], aq = bq[h], av = bv[h];
        for (int c = 0; c < 64; ++c) {
            float xv = xs[t * 64 + c];
            ak += xv * Wk[c * 64 + h]; aq += xv * Wq[c * 64 + h]; av += xv * Wv[c * 64 + h];
        }
        ks[idx] = ak; qs[idx] = aq; vs[idx] = av;
    }
    __syncthreads();
    if (tid < 128) {                       // aw[b][t][s] = q.k / sqrt(64)
        int b = tid >> 6, t = (tid >> 3) & 7, s2 = tid & 7;
        float a = 0.f;
        for (int h = 0; h < 64; ++h) a += qs[(b * 8 + t) * 64 + h] * ks[(b * 8 + s2) * 64 + h];
        aw[tid] = a * 0.125f;
    }
    __syncthreads();
    if (tid < 16) {                        // softmax over s
        float mx = -3.4e38f;
        for (int s2 = 0; s2 < 8; ++s2) mx = fmaxf(mx, aw[tid * 8 + s2]);
        float se = 0.f;
        for (int s2 = 0; s2 < 8; ++s2) { float e = expf(aw[tid * 8 + s2] - mx); aw[tid * 8 + s2] = e; se += e; }
        float inv = 1.f / se;
        for (int s2 = 0; s2 < 8; ++s2) aw[tid * 8 + s2] *= inv;
    }
    __syncthreads();
    for (int idx = tid; idx < 1024; idx += 256) {  // t1 = aw @ v
        int t = idx >> 6, h = idx & 63; int b = t >> 3;
        float a = 0.f;
        for (int s2 = 0; s2 < 8; ++s2) a += aw[t * 8 + s2] * vs[(b * 8 + s2) * 64 + h];
        t1[idx] = a;
    }
    __syncthreads();
    for (int idx = tid; idx < 1024; idx += 256) {  // qs = relu(t1 @ Wo1 + bo1)
        int t = idx >> 6, h = idx & 63;
        float a = bo1[h];
        for (int c = 0; c < 64; ++c) a += t1[t * 64 + c] * Wo1[c * 64 + h];
        qs[idx] = fmaxf(a, 0.f);
    }
    __syncthreads();
    for (int idx = tid; idx < 1024; idx += 256) {  // ks = xs + qs @ Wo2 + bo2
        int t = idx >> 6, h = idx & 63;
        float a = bo2[h];
        for (int c = 0; c < 64; ++c) a += qs[t * 64 + c] * Wo2[c * 64 + h];
        ks[idx] = xs[idx] + a;
    }
    __syncthreads();
    if (tid < 16) {                        // layernorm per token
        float m = 0.f;
        for (int h = 0; h < 64; ++h) m += ks[tid * 64 + h];
        m *= (1.f / 64.f);
        float v = 0.f;
        for (int h = 0; h < 64; ++h) { float d = ks[tid * 64 + h] - m; v += d * d; }
        v *= (1.f / 64.f);
        float inv = rsqrtf(v + EPSV);
        for (int h = 0; h < 64; ++h) ks[tid * 64 + h] = (ks[tid * 64 + h] - m) * inv * gf[h] + bff[h];
    }
    __syncthreads();
    if (tid < 128) {                       // max over s -> (B, 64)
        int b = tid >> 6, h = tid & 63;
        float m = -3.4e38f;
        for (int s2 = 0; s2 < 8; ++s2) m = fmaxf(m, ks[(b * 8 + s2) * 64 + h]);
        out[tid] = m;
    }
}

// =======================================================================
extern "C" void kernel_launch(void* const* d_in, const int* in_sizes, int n_in,
                              void* d_out, int out_size, void* d_ws, size_t ws_size,
                              hipStream_t stream) {
    (void)in_sizes; (void)n_in; (void)out_size; (void)ws_size;
    // jax pytree flatten order (sorted dict keys): params{W0,b0,bf,enc{...},gf,layers[...]}, points
    const float* W0  = (const float*)d_in[0];
    const float* b0  = (const float*)d_in[1];
    const float* bf_ = (const float*)d_in[2];
    const float* eWk = (const float*)d_in[3];
    const float* eWo1= (const float*)d_in[4];
    const float* eWo2= (const float*)d_in[5];
    const float* eWq = (const float*)d_in[6];
    const float* eWv = (const float*)d_in[7];
    const float* ebk = (const float*)d_in[8];
    const float* ebo1= (const float*)d_in[9];
    const float* ebo2= (const float*)d_in[10];
    const float* ebq = (const float*)d_in[11];
    const float* ebv = (const float*)d_in[12];
    const float* gf  = (const float*)d_in[13];
    // layer base 14 + 16*l, order: Wa,Wk,Wo,Wpd,Wpe,Wq,Wv,ba,bk,bln,bo,bpd,bpe,bq,bv,g
    const float* pts = (const float*)d_in[46];

    char* p = (char*)d_ws;
    auto alloc = [&](size_t bytes) -> void* {
        void* r = (void*)p; p += (bytes + 255) & ~(size_t)255; return r;
    };
    float*  x    = (float*)alloc((size_t)NP * HID * 4);
    __bf16* xb   = (__bf16*)alloc((size_t)NP * HID * 2);
    int*    nbr  = (int*)alloc((size_t)NP * KNB * 4);
    float*  rel  = (float*)alloc((size_t)NP * KNB * 3 * 4);
    float*  qb   = (float*)alloc((size_t)NP * HID * 4);
    float*  kb   = (float*)alloc((size_t)NP * HID * 4);
    float*  vb   = (float*)alloc((size_t)NP * HID * 4);
    float*  peb  = (float*)alloc((size_t)NP * HID * 4);
    float*  ao   = (float*)alloc((size_t)NP * HID * 4);
    __bf16* ab   = (__bf16*)alloc((size_t)NP * HID * 2);
    float*  ob   = (float*)alloc((size_t)NP * HID * 4);
    __bf16* wqb  = (__bf16*)alloc(HID * HID * 2);
    __bf16* wkb  = (__bf16*)alloc(HID * HID * 2);
    __bf16* wvb  = (__bf16*)alloc(HID * HID * 2);
    __bf16* wob  = (__bf16*)alloc(HID * HID * 2);
    float*  xp   = (float*)alloc(BB * SS * HID * 4);

    knn_kernel<<<64, 256, 0, stream>>>(pts, nbr);
    embed_kernel<<<(NP * HID) / 256, 256, 0, stream>>>(pts, W0, b0, x, xb);
    rel_kernel<<<(NP * KNB) / 256, 256, 0, stream>>>(pts, nbr, rel);

    for (int l = 0; l < 2; ++l) {
        int base = 14 + 16 * l;
        const float* Wa  = (const float*)d_in[base + 0];
        const float* Wkl = (const float*)d_in[base + 1];
        const float* Wo  = (const float*)d_in[base + 2];
        const float* Wpd = (const float*)d_in[base + 3];
        const float* Wpe = (const float*)d_in[base + 4];
        const float* Wql = (const float*)d_in[base + 5];
        const float* Wvl = (const float*)d_in[base + 6];
        const float* ba  = (const float*)d_in[base + 7];
        const float* bkl = (const float*)d_in[base + 8];
        const float* bln = (const float*)d_in[base + 9];
        const float* bo  = (const float*)d_in[base + 10];
        const float* bpd = (const float*)d_in[base + 11];
        const float* bpe = (const float*)d_in[base + 12];
        const float* bql = (const float*)d_in[base + 13];
        const float* bvl = (const float*)d_in[base + 14];
        const float* g   = (const float*)d_in[base + 15];

        pack_w_kernel<<<16, 256, 0, stream>>>(Wql, wqb);
        pack_w_kernel<<<16, 256, 0, stream>>>(Wkl, wkb);
        pack_w_kernel<<<16, 256, 0, stream>>>(Wvl, wvb);
        pack_w_kernel<<<16, 256, 0, stream>>>(Wo,  wob);

        gemm_bf16_kernel<<<NP / 16, 128, 0, stream>>>(xb, wqb, bql, qb);
        gemm_bf16_kernel<<<NP / 16, 128, 0, stream>>>(xb, wkb, bkl, kb);
        gemm_bf16_kernel<<<NP / 16, 128, 0, stream>>>(xb, wvb, bvl, vb);
        pe_kernel<<<(NP * HID) / 256, 256, 0, stream>>>(pts, Wpe, bpe, peb);

        attn_kernel<<<NP, 64, 0, stream>>>(qb, kb, vb, peb, rel, nbr, Wpd, bpd, Wa, ba, ao);

        cast_bf16_kernel<<<(NP * HID) / 256, 256, 0, stream>>>(ao, ab, NP * HID);
        gemm_bf16_kernel<<<NP / 16, 128, 0, stream>>>(ab, wob, bo, ob);

        resln_kernel<<<NP, 64, 0, stream>>>(x, ob, g, bln, xb);
    }

    maxpool_kernel<<<BB * SS, 64, 0, stream>>>(x, xp);
    final_kernel<<<1, 256, 0, stream>>>(xp, eWk, ebk, eWq, ebq, eWv, ebv,
                                        eWo1, ebo1, eWo2, ebo2, gf, bf_,
                                        (float*)d_out);
}
